// DifferentiableADF_4243427688499
// MI455X (gfx1250) — compile-verified, matrix operand order checked by reading, not run on hardware
//
#include <hip/hip_runtime.h>
#include <hip/hip_bf16.h>
#include <math.h>

typedef __attribute__((ext_vector_type(16))) _Float16 v16h;
typedef __attribute__((ext_vector_type(8)))  float    v8f;

#define NFRAMES   8
#define NATOMS    512
#define NANGLES   500000
#define NBINS     180
#define NTILES    12                 // 12*16 = 192 >= 180 bins
#define PADBINS   (NTILES * 16)
#define BIN_STEP  (180.0f / 179.0f)  // linspace(0,180,180) spacing
#define XYZ_FLOATS (NFRAMES * NATOMS * 3)
#define RAD2DEG   57.29577951308232f
// exp(-0.5*d^2) == exp2(C2 * d^2), C2 = -0.5 * log2(e)
#define C2        (-0.72134752044448169f)
#define N2C2      (1.4426950408889634f)   // -2 * C2 = log2(e)

__global__ void adf_zero_ws(float* ws) {
    int i = blockIdx.x * blockDim.x + threadIdx.x;
    if (i < PADBINS) ws[i] = 0.0f;
}

__global__ __launch_bounds__(256)
void adf_main(const float* __restrict__ xyz,
              const int*   __restrict__ angle_list,
              float*       __restrict__ ws) {
    // Stage the whole xyz table into LDS (48KB << 320KB/WGP).
    __shared__ float sxyz[XYZ_FLOATS];
    for (int i = threadIdx.x; i < XYZ_FLOATS; i += blockDim.x) sxyz[i] = xyz[i];
    __syncthreads();

    const int lane    = threadIdx.x & 31;
    const int waveId  = blockIdx.x * (blockDim.x >> 5) + (threadIdx.x >> 5);
    const int nWaves  = gridDim.x * (blockDim.x >> 5);
    const int nChunks = NANGLES / 32;   // 15625 exactly, no tail

    // A = ones(16x32) f16 : layout-invariant, makes D[m][n] = sum_k B[k][n]
    v16h ones;
    #pragma unroll
    for (int k = 0; k < 16; ++k) ones[k] = (_Float16)1.0f;

    v8f acc[NTILES];
    #pragma unroll
    for (int t = 0; t < NTILES; ++t)
        acc[t] = (v8f){0.f, 0.f, 0.f, 0.f, 0.f, 0.f, 0.f, 0.f};

    const float binCol  = (float)(lane & 15) * BIN_STEP; // this lane's bin column
    const int   selBase = lane & 16;  // lanes 0-15 need angles 0-15; 16-31 need 16-31

    for (int c = waveId; c < nChunks; c += nWaves) {
        // --- one angle per lane ---
        const int4 rec = ((const int4*)angle_list)[c * 32 + lane]; // f, i, center, j
        const int   bf = rec.x * (NATOMS * 3);
        const float* pC = &sxyz[bf + rec.z * 3];
        const float* pI = &sxyz[bf + rec.y * 3];
        const float* pJ = &sxyz[bf + rec.w * 3];
        const float v1x = pI[0] - pC[0], v1y = pI[1] - pC[1], v1z = pI[2] - pC[2];
        const float v2x = pJ[0] - pC[0], v2y = pJ[1] - pC[1], v2z = pJ[2] - pC[2];
        const float dt = v1x * v2x + v1y * v2y + v1z * v2z;
        const float n1 = v1x * v1x + v1y * v1y + v1z * v1z;
        const float n2 = v2x * v2x + v2y * v2y + v2z * v2z;
        float cosA = dt * rsqrtf(n1 * n2);
        cosA = fminf(1.0f, fmaxf(-1.0f, cosA));
        const float ang = acosf(cosA) * RAD2DEG;

        // Broadcast the 16 angles this lane's B-column needs, and hoist the
        // per-angle terms of C2*(a-off)^2 = (-2C2*a)*off + C2*a^2 + C2*off^2.
        float u[16], vv[16];
        #pragma unroll
        for (int k = 0; k < 16; ++k) {
            const float a = __shfl(ang, selBase + k, 32);
            u[k]  = C2 * a * a;
            vv[k] = N2C2 * a;
        }

        // --- 12 bin tiles: build B = gaussian weights, accumulate via WMMA ---
        v16h bb[2];   // double-buffered B: avoid WAR on the regs WMMA just read
        #pragma unroll
        for (int t = 0; t < NTILES; ++t) {
            const float off = binCol + (float)(t * 16) * BIN_STEP;
            const float q   = C2 * off * off;
            v16h& b = bb[t & 1];
            #pragma unroll
            for (int k = 0; k < 16; ++k) {
                // arg = C2*(a-off)^2 via 1 fma + 1 add; v_exp_f32 is native 2^x
                b[k] = (_Float16)__builtin_amdgcn_exp2f(
                           __builtin_fmaf(vv[k], off, u[k]) + q);
            }
            // D = ones(16x32) * B(32x16) + C : column sums of B into every row
            acc[t] = __builtin_amdgcn_wmma_f32_16x16x32_f16(
                false, ones, false, b, (short)0, acc[t], false, false);
        }
    }

    // C layout: VGPR0, lanes 0..15 = row M=0, N=lane -> per-bin wave partials.
    if (lane < 16) {
        #pragma unroll
        for (int t = 0; t < NTILES; ++t) {
            const int bin = t * 16 + lane;
            if (bin < NBINS) atomicAdd(&ws[bin], acc[t][0]);
        }
    }
}

__global__ void adf_finalize(const float* __restrict__ ws, float* __restrict__ out) {
    __shared__ float red[256];
    const int b = threadIdx.x;
    const float v = (b < NBINS) ? ws[b] : 0.0f;
    red[b] = v;
    __syncthreads();
    #pragma unroll
    for (int s = 128; s > 0; s >>= 1) {
        if (b < s) red[b] += red[b + s];
        __syncthreads();
    }
    if (b < NBINS) out[b] = v / red[0];
}

extern "C" void kernel_launch(void* const* d_in, const int* in_sizes, int n_in,
                              void* d_out, int out_size, void* d_ws, size_t ws_size,
                              hipStream_t stream) {
    const float* xyz        = (const float*)d_in[0];
    const int*   angle_list = (const int*)d_in[1];
    float*       out        = (float*)d_out;
    float*       ws         = (float*)d_ws;

    adf_zero_ws<<<1, 256, 0, stream>>>(ws);
    adf_main<<<240, 256, 0, stream>>>(xyz, angle_list, ws);
    adf_finalize<<<1, 256, 0, stream>>>(ws, out);
}
